// XLSTMDecoder_69183333204518
// MI455X (gfx1250) — compile-verified, matrix operand order checked by reading, not run on hardware
//
#include <hip/hip_runtime.h>
#include <hip/hip_bf16.h>
#include <cstdint>

typedef __bf16 bf16;
typedef __attribute__((ext_vector_type(16))) __bf16 v16bf;
typedef __attribute__((ext_vector_type(8)))  __bf16 v8bf;
typedef __attribute__((ext_vector_type(8)))  float  v8f;

#define B_    32
#define T_    1024
#define C_    512
#define D_    768
#define NP_   253
#define ST_   4
#define NH_   4
#define DH_   192
#define NCLS_ 41
#define LNUM_ 5
#define M2_   (B_ * NP_)     // 8096
#define EPS_  1e-5f

union F8u  { v8f  v; float e[8];  };
union F16u { v16bf v; v8bf h[2]; bf16 e[16]; };
union BF8u { v8bf v; bf16 e[8]; };

// k-index inside a 16-bit WMMA A/B fragment (ISA 7.12.2 layout)
__device__ __forceinline__ int kidx(int e, int half) {
  return (e & 7) + half * 8 + (e >> 3) * 16;
}

// ---------------------------------------------------------------------------
// Prep kernels: fp32 -> bf16 weight/activation conversion & repacking
// ---------------------------------------------------------------------------
__global__ void k_convert(const float* __restrict__ s, bf16* __restrict__ d, long n) {
  long i = (long)blockIdx.x * 256 + threadIdx.x;
  if (i < n) d[i] = (bf16)s[i];
}

__global__ void k_dayprep(const float* __restrict__ dayW, const float* __restrict__ dayb,
                          const int* __restrict__ idx, bf16* __restrict__ Wb, float* __restrict__ bb) {
  int day = idx[0];
  long i = (long)blockIdx.x * 256 + threadIdx.x;
  if (i < (long)C_ * C_) Wb[i] = (bf16)dayW[(size_t)day * C_ * C_ + i];
  if (i < C_)            bb[i] = dayb[(size_t)day * C_ + i];
}

// in_proj_W (768, 7168) -> (7168, 768) bf16
__global__ void k_inprojprep(const float* __restrict__ s, bf16* __restrict__ d) {
  long i = (long)blockIdx.x * 256 + threadIdx.x;
  if (i < (long)7168 * 768) {
    int k = i / 768, n = i % 768;
    d[i] = (bf16)s[(size_t)n * 7168 + k];
  }
}

// gate_W (L,4,NH,192,192) -> [l][h][pair][k=192][n=384] bf16  (pair=g>>1, col=(g&1)*192+e)
__global__ void k_gateprep(const float* __restrict__ s, bf16* __restrict__ d) {
  long i = (long)blockIdx.x * 256 + threadIdx.x;
  if (i >= (long)LNUM_ * 4 * NH_ * DH_ * DH_) return;
  int e = i % 192;
  int dd = (i / 192) % 192;
  int h = (i / 36864) % 4;
  int g = (i / 147456) % 4;
  int l = i / 589824;
  size_t dst = ((((size_t)l * 4 + h) * 2 + (g >> 1)) * 192 + dd) * 384 + (g & 1) * 192 + e;
  d[dst] = (bf16)s[i];
}

// rec_W (L,NH,192,4,192) -> fragment-major bf16: per (l,h): frag f=(kt*4+g)*12+ntile,
// element addr = (f*32 + lane)*16 + e ; value = R[k(kidx), n=ntile*16+lane15] of gate g
__global__ void k_recprep(const float* __restrict__ s, bf16* __restrict__ d) {
  long i = (long)blockIdx.x * 256 + threadIdx.x;
  if (i >= (long)LNUM_ * NH_ * 192 * 768) return;
  int e2   = i % 16;
  int lane = (i / 16) % 32;
  int f    = (i / 512) % 288;
  int lh   = i / 147456;
  int ntile = f % 12;
  int g     = (f / 12) % 4;
  int kt    = f / 48;
  int k = kt * 32 + kidx(e2, lane >> 4);
  int n = ntile * 16 + (lane & 15);
  d[i] = (bf16)s[(((size_t)lh * 192 + k) * 4 + g) * 192 + n];
}

// out_W (41,768) -> padded (768, 64) bf16, zero-filled
__global__ void k_outprep(const float* __restrict__ s, bf16* __restrict__ d) {
  int i = blockIdx.x * 256 + threadIdx.x;
  if (i < 768 * 64) {
    int k = i / 64, n = i % 64;
    d[i] = (n < NCLS_) ? (bf16)s[(size_t)n * 768 + k] : (bf16)0.0f;
  }
}

// ---------------------------------------------------------------------------
// Generic bf16 WMMA GEMM, LDS tiles stored fragment-major.
// 256 threads, 8 waves (2 m x 4 n), block tile 64 x BN, BK = 64.
// AMODE: 0 = plain row-major (lda, a_coloff), 1 = patchify gather from (B,T,C)
// EPI:   0 = f32, 1 = f32+bias, 2 = bias+softsign->bf16, 3 = scan-fragment bf16
// ---------------------------------------------------------------------------
#define GBM 64
#define GBK 64

template<int AMODE>
__device__ __forceinline__ bf16 fetchA(const bf16* __restrict__ A, long lda, int coloff,
                                       int row, int k, int M) {
  if (row >= M) row = M - 1;
  if (AMODE == 0) {
    return A[(size_t)row * lda + coloff + k];
  } else {
    int b = row / NP_, p = row % NP_;
    int t = p * ST_ + (k >> 9);
    int c = k & 511;
    return A[((size_t)b * T_ + t) * C_ + c];
  }
}

template<int BN, int AMODE, int EPI>
__global__ __launch_bounds__(256) void gemm_kernel(
    const bf16* __restrict__ A, long lda, int a_coloff,
    const bf16* __restrict__ Bw, long ldb,
    float* __restrict__ Cf, bf16* __restrict__ Cb, long ldc, int c_coloff,
    const float* __restrict__ bias, int M, int N, int Kd, int p_h, int p_pair)
{
  constexpr int NFRB = BN / 16;   // B n-tiles per block
  constexpr int NT   = BN / 64;   // n-tiles per wave
  __shared__ __align__(32) bf16 AsF[4 * 2 * 32 * 16];
  __shared__ __align__(32) bf16 BsF[NFRB * 2 * 32 * 16];
  int t = threadIdx.x;
  int lane = t & 31, wid = t >> 5;
  int wm = wid & 1, wn = wid >> 1;
  int half = lane >> 4, l15 = lane & 15;
  int blockM = blockIdx.y * GBM;
  int blockN = blockIdx.x * BN;

  F8u acc[2][NT];
#pragma unroll
  for (int mi = 0; mi < 2; ++mi)
#pragma unroll
    for (int ni = 0; ni < NT; ++ni)
#pragma unroll
      for (int r = 0; r < 8; ++r) acc[mi][ni].e[r] = 0.f;

  for (int k0 = 0; k0 < Kd; k0 += GBK) {
    __syncthreads();
    { // A tile -> fragment-major LDS
      int r = t >> 2, c0 = (t & 3) * 16;
      int grow = blockM + r;
      int mt = r >> 4, l15m = r & 15;
      bf16 tmpA[16];
#pragma unroll
      for (int j = 0; j < 16; ++j)
        tmpA[j] = fetchA<AMODE>(A, lda, a_coloff, grow, k0 + c0 + j, M);
#pragma unroll
      for (int j = 0; j < 16; ++j) {
        int c = c0 + j;
        int kk = c >> 5, crm = c & 31;
        int hf = (crm >> 3) & 1;
        int e = (crm & 7) + ((crm >> 4) << 3);
        AsF[(((mt * 2 + kk) * 32) + hf * 16 + l15m) * 16 + e] = tmpA[j];
      }
    }
    { // B tile -> fragment-major LDS
      constexpr int PER = GBK * BN / 256;
      int idx0 = t * PER;
      int rB = idx0 / BN, cB0 = idx0 % BN;
      const bf16* src = Bw + (size_t)(k0 + rB) * ldb + blockN + cB0;
      if (k0 + GBK < Kd) __builtin_prefetch(src + (size_t)GBK * ldb, 0, 0);
      int kkB = rB >> 5, krm = rB & 31;
      int hfB = (krm >> 3) & 1;
      int eB = (krm & 7) + ((krm >> 4) << 3);
      bf16 tmpB[PER];
#pragma unroll
      for (int j = 0; j < PER; ++j) tmpB[j] = src[j];
#pragma unroll
      for (int j = 0; j < PER; ++j) {
        int c = cB0 + j;
        int nt = c >> 4, l15n = c & 15;
        BsF[((nt * 2 + kkB) * 32 + hfB * 16 + l15n) * 16 + eB] = tmpB[j];
      }
    }
    __syncthreads();
#pragma unroll
    for (int kk = 0; kk < 2; ++kk) {
      F16u af[2];
#pragma unroll
      for (int mi = 0; mi < 2; ++mi)
        af[mi].v = *(const v16bf*)&AsF[(((wm * 2 + mi) * 2 + kk) * 32 + lane) * 16];
#pragma unroll
      for (int ni = 0; ni < NT; ++ni) {
        F16u bfr;
        bfr.v = *(const v16bf*)&BsF[(((wn * NT + ni) * 2 + kk) * 32 + lane) * 16];
#pragma unroll
        for (int mi = 0; mi < 2; ++mi)
          acc[mi][ni].v = __builtin_amdgcn_wmma_f32_16x16x32_bf16(
              false, af[mi].v, false, bfr.v, (short)0, acc[mi][ni].v, false, false);
      }
    }
  }
  // epilogue
#pragma unroll
  for (int mi = 0; mi < 2; ++mi)
#pragma unroll
    for (int ni = 0; ni < NT; ++ni)
#pragma unroll
      for (int r = 0; r < 8; ++r) {
        int mrow = blockM + wm * 32 + mi * 16 + r + half * 8;
        int ncol = blockN + wn * (16 * NT) + ni * 16 + l15;
        if (mrow < M && ncol < N) {
          float v = acc[mi][ni].e[r];
          if (EPI == 2) {
            v += bias[ncol];
            v = v / (1.0f + fabsf(v));
            Cb[(size_t)mrow * ldc + c_coloff + ncol] = (bf16)v;
          } else if (EPI == 3) {
            // scan-fragment layout: [s][h][bg][gg][et][lane][r], bf16
            int b = mrow / NP_, s = mrow % NP_;
            int bg = b >> 4, bl = b & 15;
            int rr = bl & 7, hf2 = bl >> 3;
            int gg = p_pair * 2 + (ncol / 192);
            int e = ncol % 192;
            int et = e >> 4, l15e = e & 15;
            int lane2 = hf2 * 16 + l15e;
            size_t slot = ((((size_t)s * NH_ + p_h) * 2 + bg) * 12288)
                        + ((size_t)(gg * 12 + et) * 32 + lane2) * 8 + rr;
            Cb[slot] = (bf16)v;
          } else {
            if (EPI == 1) v += bias[ncol];
            Cf[(size_t)mrow * ldc + c_coloff + ncol] = v;
          }
        }
      }
}

// ---------------------------------------------------------------------------
// LayerNorm over D=768 per row: one wave per row, writes f32 + bf16
// ---------------------------------------------------------------------------
__global__ __launch_bounds__(256) void k_layernorm(const float* __restrict__ x,
                                                   const float* __restrict__ w,
                                                   float* __restrict__ yf, bf16* __restrict__ yb,
                                                   int rows) {
  int wid = threadIdx.x >> 5, lane = threadIdx.x & 31;
  int row = blockIdx.x * 8 + wid;
  if (row >= rows) return;
  const float* xr = x + (size_t)row * D_;
  float v[24], s = 0.f, sq = 0.f;
#pragma unroll
  for (int q = 0; q < 24; ++q) { float u = xr[lane + 32 * q]; v[q] = u; s += u; sq += u * u; }
#pragma unroll
  for (int m = 16; m; m >>= 1) { s += __shfl_xor(s, m, 32); sq += __shfl_xor(sq, m, 32); }
  float mu = s * (1.f / D_);
  float var = sq * (1.f / D_) - mu * mu;
  float rs = rsqrtf(var + EPS_);
#pragma unroll
  for (int q = 0; q < 24; ++q) {
    int e = lane + 32 * q;
    float o = (v[q] - mu) * rs * w[e];
    yf[(size_t)row * D_ + e] = o;
    yb[(size_t)row * D_ + e] = (bf16)o;
  }
}

// ---------------------------------------------------------------------------
// Causal depthwise conv (k=4) + SiLU, output bf16
// ---------------------------------------------------------------------------
__global__ void k_conv(const float* __restrict__ xin, const float* __restrict__ cw,
                       const float* __restrict__ cb, bf16* __restrict__ xc, long total) {
  long i = (long)blockIdx.x * 256 + threadIdx.x;
  if (i >= total) return;
  int d = i % D_;
  long row = i / D_;
  int p = row % NP_;
  long rb = row - p;
  float a = cb[d];
#pragma unroll
  for (int j = 0; j < 4; ++j) {
    int pp = p - 3 + j;
    if (pp >= 0) a += cw[d * 4 + j] * xin[(size_t)(rb + pp) * D_ + d];
  }
  float sg = 1.f / (1.f + __expf(-a));
  xc[i] = (bf16)(a * sg);
}

// ---------------------------------------------------------------------------
// sLSTM scan: 8 blocks = 4 heads x 2 batch-groups of 16; 384 threads (12 waves),
// one 16-wide e-tile per wave. Per step: raw = gates + h @ R + cell_b via WMMA.
// Gate fragments async-copied global->LDS (double-buffered) overlapping compute.
// h double-buffered bf16 in LDS; c/n/m in registers (no spill).
// ---------------------------------------------------------------------------
__global__ __launch_bounds__(384) void k_scan(const bf16* __restrict__ g,
                                              const bf16* __restrict__ Rl,
                                              const float* __restrict__ cbl,
                                              float* __restrict__ y) {
  int hh = blockIdx.x >> 1, bg = blockIdx.x & 1;
  int t = threadIdx.x, lane = t & 31, et = t >> 5;
  int half = lane >> 4, l15 = lane & 15;

  __shared__ __align__(16) bf16 hS[2][16][200];
  __shared__ __align__(16) bf16 gS[2][12288];

  for (int i = t; i < 2 * 16 * 200; i += 384) {
    int buf = i / 3200, rem = i % 3200;
    hS[buf][rem / 200][rem % 200] = (bf16)0.0f;
  }

  const bf16* Rh = Rl + (size_t)hh * 147456;
  const float* cb = cbl + (size_t)hh * 768;
  float cbv[4];
#pragma unroll
  for (int gg = 0; gg < 4; ++gg) cbv[gg] = cb[gg * 192 + et * 16 + l15];

  float cS[8], nS[8], mS[8];
#pragma unroll
  for (int r = 0; r < 8; ++r) { cS[r] = 0.f; nS[r] = 0.f; mS[r] = 0.f; }

  auto issue_async = [&](int buf, int s) {
    const bf16* slab = g + ((((size_t)s * NH_ + hh) * 2 + bg) * 12288);
    unsigned lbase = (unsigned)(uintptr_t)&gS[buf][0];
#pragma unroll
    for (int j = 0; j < 4; ++j) {
      int c = t + 384 * j;                 // 16-byte chunk id (1536 total)
      unsigned loff = lbase + c * 16;
      const bf16* gp = slab + c * 8;
      asm volatile("global_load_async_to_lds_b128 %0, %1, off"
                   :: "v"(loff), "v"(gp) : "memory");
    }
  };

  issue_async(0, 0);
  asm volatile("s_wait_asynccnt 0" ::: "memory");
  __syncthreads();

  for (int s = 0; s < NP_; ++s) {
    if (s + 1 < NP_) issue_async((s + 1) & 1, s + 1);  // overlap with compute
    F8u acc[4];
#pragma unroll
    for (int gg = 0; gg < 4; ++gg) {
      BF8u g8;
      g8.v = *(const v8bf*)&gS[s & 1][((gg * 12 + et) * 32 + lane) * 8];
#pragma unroll
      for (int r = 0; r < 8; ++r) acc[gg].e[r] = (float)g8.e[r] + cbv[gg];
    }
#pragma unroll
    for (int kt = 0; kt < 6; ++kt) {
      F16u af;
      af.h[0] = *(const v8bf*)&hS[s & 1][l15][kt * 32 + half * 8];
      af.h[1] = *(const v8bf*)&hS[s & 1][l15][kt * 32 + 16 + half * 8];
#pragma unroll
      for (int gg = 0; gg < 4; ++gg) {
        F16u bfr;
        bfr.v = *(const v16bf*)(Rh + ((size_t)((kt * 4 + gg) * 12 + et) * 32 + lane) * 16);
        acc[gg].v = __builtin_amdgcn_wmma_f32_16x16x32_bf16(
            false, af.v, false, bfr.v, (short)0, acc[gg].v, false, false);
      }
    }
#pragma unroll
    for (int r = 0; r < 8; ++r) {
      int bl = r + half * 8;
      float ir = acc[0].e[r], fr = acc[1].e[r], zr = acc[2].e[r], orw = acc[3].e[r];
      float ls = fminf(fr, 0.f) - __logf(1.f + __expf(-fabsf(fr)));
      float lf = mS[r] + ls;
      float mn = (s == 0) ? ir : fmaxf(ir, lf);
      float ig = __expf(ir - mn);
      float fg = __expf(lf - mn);
      float cn = fg * cS[r] + ig * tanhf(zr);
      float nn = fg * nS[r] + ig;
      float yv = (1.f / (1.f + __expf(-orw))) * cn / nn;
      cS[r] = cn; nS[r] = nn; mS[r] = mn;
      hS[(s + 1) & 1][bl][et * 16 + l15] = (bf16)yv;
      y[(((size_t)(bg * 16 + bl) * NP_ + s) * NH_ + hh) * DH_ + et * 16 + l15] = yv;
    }
    asm volatile("s_wait_asynccnt 0" ::: "memory");
    __syncthreads();
  }
}

// ---------------------------------------------------------------------------
// GroupNorm per (row, head) over 192 + residual add into x
// ---------------------------------------------------------------------------
__global__ __launch_bounds__(256) void k_groupnorm_res(const float* __restrict__ y,
                                                       const float* __restrict__ w,
                                                       float* __restrict__ x, int rows) {
  int wid = threadIdx.x >> 5, lane = threadIdx.x & 31;
  int idx = blockIdx.x * 8 + wid;
  if (idx >= rows * NH_) return;
  int row = idx >> 2, h = idx & 3;
  const float* yr = y + ((size_t)row * NH_ + h) * DH_;
  float v[6], s = 0.f, sq = 0.f;
#pragma unroll
  for (int q = 0; q < 6; ++q) { float u = yr[lane + 32 * q]; v[q] = u; s += u; sq += u * u; }
#pragma unroll
  for (int m = 16; m; m >>= 1) { s += __shfl_xor(s, m, 32); sq += __shfl_xor(sq, m, 32); }
  float mu = s * (1.f / DH_);
  float var = sq * (1.f / DH_) - mu * mu;
  float rs = rsqrtf(var + EPS_);
#pragma unroll
  for (int q = 0; q < 6; ++q) {
    int e = lane + 32 * q;
    x[(size_t)row * D_ + h * DH_ + e] += (v[q] - mu) * rs * w[h * DH_ + e];
  }
}

// ---------------------------------------------------------------------------
// Host launch
// ---------------------------------------------------------------------------
extern "C" void kernel_launch(void* const* d_in, const int* in_sizes, int n_in,
                              void* d_out, int out_size, void* d_ws, size_t ws_size,
                              hipStream_t stream) {
  const float* features  = (const float*)d_in[0];
  const int*   dayidx    = (const int*)d_in[1];
  const float* day_W     = (const float*)d_in[2];
  const float* day_b     = (const float*)d_in[3];
  const float* in_proj_W = (const float*)d_in[4];
  const float* in_proj_b = (const float*)d_in[5];
  const float* ln_w      = (const float*)d_in[6];
  const float* conv_W    = (const float*)d_in[7];
  const float* conv_b    = (const float*)d_in[8];
  const float* gate_W    = (const float*)d_in[9];
  const float* rec_W     = (const float*)d_in[10];
  const float* cell_b    = (const float*)d_in[11];
  const float* gn_w      = (const float*)d_in[12];
  const float* post_ln_w = (const float*)d_in[13];
  const float* out_W     = (const float*)d_in[14];
  const float* out_b     = (const float*)d_in[15];
  float* out = (float*)d_out;

  char* ws = (char*)d_ws;
  size_t off = 0;
  auto alloc = [&](size_t bytes) { size_t o = off; off = (off + bytes + 255) & ~(size_t)255; return o; };

  const long NFEAT = (long)B_ * T_ * C_;        // 16,777,216
  const long NSEQ  = (long)M2_ * D_;            //  6,217,728

  bf16*  fbf     = (bf16*) (ws + alloc(NFEAT * 2));
  bf16*  xbuf    = (bf16*) (ws + alloc(NFEAT * 2));
  bf16*  dayWb   = (bf16*) (ws + alloc((size_t)C_ * C_ * 2));
  float* daybias = (float*)(ws + alloc(C_ * 4));
  bf16*  inpWb   = (bf16*) (ws + alloc((size_t)7168 * 768 * 2));
  bf16*  gwb     = (bf16*) (ws + alloc((size_t)LNUM_ * NH_ * 2 * 192 * 384 * 2));
  bf16*  recWb   = (bf16*) (ws + alloc((size_t)LNUM_ * NH_ * 147456 * 2));
  bf16*  outWb   = (bf16*) (ws + alloc((size_t)768 * 64 * 2));
  float* xseq    = (float*)(ws + alloc(NSEQ * 4));
  float* xin     = (float*)(ws + alloc(NSEQ * 4));
  bf16*  xinbf   = (bf16*) (ws + alloc(NSEQ * 2));
  bf16*  xcbf    = (bf16*) (ws + alloc(NSEQ * 2));
  bf16*  gbuf    = (bf16*) (ws + alloc((size_t)M2_ * 3072 * 2));   // scan-fragment bf16
  float* ybuf    = (float*)(ws + alloc(NSEQ * 4));

  // --- weight / input prep ---
  k_convert   <<<(NFEAT + 255) / 256, 256, 0, stream>>>(features, fbf, NFEAT);
  k_dayprep   <<<((long)C_ * C_ + 255) / 256, 256, 0, stream>>>(day_W, day_b, dayidx, dayWb, daybias);
  k_inprojprep<<<((long)7168 * 768 + 255) / 256, 256, 0, stream>>>(in_proj_W, inpWb);
  k_gateprep  <<<((long)LNUM_ * 4 * NH_ * DH_ * DH_ + 255) / 256, 256, 0, stream>>>(gate_W, gwb);
  k_recprep   <<<((long)LNUM_ * NH_ * 192 * 768 + 255) / 256, 256, 0, stream>>>(rec_W, recWb);
  k_outprep   <<<(768 * 64 + 255) / 256, 256, 0, stream>>>(out_W, outWb);

  // --- day projection + softsign -> xbuf (bf16) ---
  gemm_kernel<128, 0, 2><<<dim3(C_ / 128, (B_ * T_) / GBM), 256, 0, stream>>>(
      fbf, C_, 0, dayWb, C_, nullptr, xbuf, C_, 0, daybias, B_ * T_, C_, C_, 0, 0);

  // --- patchify + in_proj -> xseq (f32) ---
  gemm_kernel<128, 1, 1><<<dim3(D_ / 128, (M2_ + GBM - 1) / GBM), 256, 0, stream>>>(
      xbuf, 0, 0, inpWb, D_, xseq, nullptr, D_, 0, in_proj_b, M2_, D_, 7168, 0, 0);

  for (int l = 0; l < LNUM_; ++l) {
    k_layernorm<<<M2_ / 8, 256, 0, stream>>>(xseq, ln_w + (size_t)l * D_, xin, xinbf, M2_);
    k_conv<<<((long)M2_ * D_ + 255) / 256, 256, 0, stream>>>(
        xin, conv_W + (size_t)l * D_ * 4, conv_b + (size_t)l * D_, xcbf, (long)M2_ * D_);
    for (int h = 0; h < NH_; ++h)
      for (int pair = 0; pair < 2; ++pair) {
        const bf16* Ain = (pair == 0) ? xcbf : xinbf;
        const bf16* Wp = gwb + (((size_t)l * NH_ + h) * 2 + pair) * 192 * 384;
        gemm_kernel<128, 0, 3><<<dim3(3, (M2_ + GBM - 1) / GBM), 256, 0, stream>>>(
            Ain, D_, h * DH_, Wp, 384, nullptr, gbuf, 0, 0,
            nullptr, M2_, 384, 192, h, pair);
      }
    k_scan<<<2 * NH_, 384, 0, stream>>>(gbuf, recWb + (size_t)l * NH_ * 147456,
                                        cell_b + (size_t)l * NH_ * 4 * DH_, ybuf);
    k_groupnorm_res<<<(M2_ * NH_) / 8, 256, 0, stream>>>(ybuf, gn_w + (size_t)l * D_, xseq, M2_);
  }

  // --- post-LN + output projection ---
  k_layernorm<<<M2_ / 8, 256, 0, stream>>>(xseq, post_ln_w, xin, xinbf, M2_);
  gemm_kernel<64, 0, 1><<<dim3(1, (M2_ + GBM - 1) / GBM), 256, 0, stream>>>(
      xinbf, D_, 0, outWb, 64, out, nullptr, NCLS_, 0, out_b, M2_, NCLS_, D_, 0, 0);

  (void)in_sizes; (void)n_in; (void)out_size; (void)ws_size;
}
